// DistributedAttention_27599459844962
// MI455X (gfx1250) — matrix operand order repaired and checked by reading, not compiled, not run on hardware
//
#include <hip/hip_runtime.h>

typedef __attribute__((ext_vector_type(16))) _Float16 v16h;
typedef __attribute__((ext_vector_type(8)))  _Float16 v8h;
typedef __attribute__((ext_vector_type(8)))  float    v8f;

#define ROWS 4096     // B*S
#define HID  1024
#define NH   16
#define HD   64
#define NQKV 3072
#define OUT0 67108864 // B*S*H*HID elements of main output

// ---------- WMMA fragment loaders (CDNA5 wave32 layouts, 16x16x32 f16) ----------

// A-matrix 16x32 (MxK), f16, row-major source with leading dim lda.
// Lane l: M = l&15. Lanes 0-15 hold K 0..15; lanes 16-31 hold K 16..31 (contiguous).
__device__ __forceinline__ v16h load_a16x32(const _Float16* A, int lda, int row0, int k0, int lane) {
    int m = lane & 15, half = lane >> 4;
    const _Float16* p = A + (size_t)(row0 + m) * lda + k0;
    v8h lo = *(const v8h*)(p + half * 16);
    v8h hi = *(const v8h*)(p + half * 16 + 8);
    v16h a;
#pragma unroll
    for (int j = 0; j < 8; ++j) { a[j] = lo[j]; a[8 + j] = hi[j]; }
    return a;
}

// B-matrix 32x16 (KxN), f16, loaded from TRANSPOSED weights Bt[n][k] with leading dim ldb.
// Lane l: N = l&15; lanes 0-15 hold K=0..15, lanes 16-31 hold K=16..31 (2 per VGPR, contiguous).
__device__ __forceinline__ v16h load_b32x16t(const _Float16* Bt, int ldb, int col0, int k0, int lane) {
    int n = lane & 15, khalf = (lane >> 4) << 4;
    return *(const v16h*)(Bt + (size_t)(col0 + n) * ldb + k0 + khalf);
}

// ---------- weight packing ----------

__global__ void k_pack_wqkv(const float* __restrict__ Wq, const float* __restrict__ Wk,
                            const float* __restrict__ Wv, _Float16* __restrict__ Wt) {
    int idx = blockIdx.x * blockDim.x + threadIdx.x;   // 0 .. 3072*1024-1
    int n = idx >> 10;                                 // fused output column
    int k = idx & 1023;                                // input dim
    const float* W = (n < 1024) ? Wq : (n < 2048 ? Wk : Wv);
    Wt[idx] = (_Float16)W[(size_t)k * 1024 + (n & 1023)];
}

__global__ void k_pack_wo(const float* __restrict__ Wo, _Float16* __restrict__ Wot) {
    int idx = blockIdx.x * blockDim.x + threadIdx.x;   // 0 .. 1024*64-1
    int n = idx >> 6, k = idx & 63;
    Wot[idx] = (_Float16)Wo[(size_t)k * 1024 + n];
}

// ---------- LayerNorm (block per row) ----------

__global__ void k_layernorm(const float* __restrict__ x, const float* __restrict__ g,
                            const float* __restrict__ b, _Float16* __restrict__ xn) {
    __shared__ float sS[256], sS2[256];
    int row = blockIdx.x, tid = threadIdx.x;
    const float* xr = x + (size_t)row * HID;
    float s = 0.f, s2 = 0.f;
    for (int c = tid; c < HID; c += 256) { float v = xr[c]; s += v; s2 += v * v; }
    sS[tid] = s; sS2[tid] = s2;
    __syncthreads();
    for (int o = 128; o > 0; o >>= 1) {
        if (tid < o) { sS[tid] += sS[tid + o]; sS2[tid] += sS2[tid + o]; }
        __syncthreads();
    }
    float mu  = sS[0]  * (1.f / HID);
    float var = sS2[0] * (1.f / HID) - mu * mu;   // population variance (jnp.var)
    float r = rsqrtf(var + 1e-6f);
    _Float16* orow = xn + (size_t)row * HID;
    for (int c = tid; c < HID; c += 256)
        orow[c] = (_Float16)((xr[c] - mu) * r * g[c] + b[c]);
}

// ---------- QKV GEMM: (4096x1024) @ (1024x3072), wave computes 16x64 tile ----------

__global__ void k_qkv_gemm(const _Float16* __restrict__ Xn, const _Float16* __restrict__ Wt,
                           const float* __restrict__ bq, const float* __restrict__ bk,
                           const float* __restrict__ bv, float* __restrict__ QKV) {
    int lane = threadIdx.x & 31;
    int wave = blockIdx.x * (blockDim.x >> 5) + (threadIdx.x >> 5);
    const int NB = NQKV / 64;            // 48 column blocks
    int mtile = wave / NB, nblk = wave % NB;
    int row0 = mtile * 16, col0 = nblk * 64;

    v8f acc[4] = {};
    for (int k0 = 0; k0 < HID; k0 += 32) {
        v16h a = load_a16x32(Xn, HID, row0, k0, lane);
#pragma unroll
        for (int t = 0; t < 4; ++t) {
            v16h bfr = load_b32x16t(Wt, HID, col0 + 16 * t, k0, lane);
            acc[t] = __builtin_amdgcn_wmma_f32_16x16x32_f16(
                false, a, false, bfr, (short)0, acc[t], false, false);
        }
    }
    int n = lane & 15, mo = (lane >> 4) * 8;
#pragma unroll
    for (int t = 0; t < 4; ++t) {
        int col = col0 + 16 * t + n;
        float bias = (col < 1024) ? bq[col] : (col < 2048 ? bk[col - 1024] : bv[col - 2048]);
#pragma unroll
        for (int j = 0; j < 8; ++j)
            QKV[(size_t)(row0 + mo + j) * NQKV + col] = acc[t][j] + bias;
    }
}

// ---------- per-position head-attention + metrics (block per position) ----------

__global__ void k_attention(const float* __restrict__ QKV, _Float16* __restrict__ attnH,
                            float* __restrict__ ent, float* __restrict__ divr,
                            float* __restrict__ gnorm) {
    __shared__ float sQ[NH * HD], sK[NH * HD], sV[NH * HD];
    __shared__ float sP[NH * NH], sA[NH * HD];
    __shared__ float sD[HD], sE[NH], sG[NH];
    int pos = blockIdx.x, tid = threadIdx.x;
    const float* base = QKV + (size_t)pos * NQKV;
    for (int i = tid; i < NH * HD; i += 256) {
        sQ[i] = base[i]; sK[i] = base[1024 + i]; sV[i] = base[2048 + i];
    }
    __syncthreads();
    {   // scores (16x16 over heads), scale 1/sqrt(64)
        int h = tid >> 4, t = tid & 15;
        float sc = 0.f;
        for (int d = 0; d < HD; ++d) sc += sQ[h * HD + d] * sK[t * HD + d];
        sP[tid] = sc * 0.125f;
    }
    __syncthreads();
    if (tid < NH) {   // row softmax
        float m = sP[tid * 16];
        for (int t = 1; t < 16; ++t) m = fmaxf(m, sP[tid * 16 + t]);
        float sum = 0.f;
        for (int t = 0; t < 16; ++t) { float e = expf(sP[tid * 16 + t] - m); sP[tid * 16 + t] = e; sum += e; }
        float inv = 1.f / sum;
        for (int t = 0; t < 16; ++t) sP[tid * 16 + t] *= inv;
    }
    __syncthreads();
    for (int i = tid; i < NH * HD; i += 256) {   // attn = probs @ V
        int h = i >> 6, d = i & 63;
        float a = 0.f;
        for (int t = 0; t < 16; ++t) a += sP[h * 16 + t] * sV[t * HD + d];
        sA[i] = a;
        attnH[(size_t)pos * (NH * HD) + i] = (_Float16)a;
    }
    __syncthreads();
    if (tid < NH) {   // per-head entropy term + grad-norm
        float e = 0.f, g = 0.f;
        for (int d = 0; d < HD; ++d) { float a = sA[tid * HD + d]; e += a * logf(a + 1e-10f); }
        for (int d = 0; d < HD; ++d) {
            float q = sQ[tid * HD + d], k = sK[tid * HD + d], v = sV[tid * HD + d];
            g += q * q + k * k + v * v;
        }
        sE[tid] = e; sG[tid] = sqrtf(g);
    }
    if (tid >= 64 && tid < 128) {   // per-dim std over heads
        int d = tid - 64;
        float m = 0.f;
        for (int h = 0; h < NH; ++h) m += sA[h * HD + d];
        m *= (1.f / NH);
        float var = 0.f;
        for (int h = 0; h < NH; ++h) { float df = sA[h * HD + d] - m; var += df * df; }
        sD[d] = sqrtf(var * (1.f / NH));
    }
    __syncthreads();
    if (tid == 0) { float e = 0.f; for (int h = 0; h < NH; ++h) e += sE[h]; ent[pos]  = -e * (1.f / NH); }
    if (tid == 1) { float g = 0.f; for (int h = 0; h < NH; ++h) g += sG[h]; gnorm[pos] = g * (1.f / NH); }
    if (tid == 2) { float s = 0.f; for (int d = 0; d < HD; ++d) s += sD[d]; divr[pos]  = s * (1.f / HD); }
}

// ---------- output GEMM: (65536x64) @ (64x1024), wave computes 16x64 tile ----------

__global__ void k_out_gemm(const _Float16* __restrict__ A, const _Float16* __restrict__ Wot,
                           const float* __restrict__ bo, float* __restrict__ out) {
    int lane = threadIdx.x & 31;
    int wave = blockIdx.x * (blockDim.x >> 5) + (threadIdx.x >> 5);
    const int NB = HID / 64;             // 16 column blocks
    int mtile = wave / NB, nblk = wave % NB;
    int row0 = mtile * 16, col0 = nblk * 64;

    v8f acc[4] = {};
#pragma unroll
    for (int k0 = 0; k0 < HD; k0 += 32) {
        v16h a = load_a16x32(A, HD, row0, k0, lane);
#pragma unroll
        for (int t = 0; t < 4; ++t) {
            v16h bfr = load_b32x16t(Wot, HD, col0 + 16 * t, k0, lane);
            acc[t] = __builtin_amdgcn_wmma_f32_16x16x32_f16(
                false, a, false, bfr, (short)0, acc[t], false, false);
        }
    }
    int n = lane & 15, mo = (lane >> 4) * 8;
#pragma unroll
    for (int t = 0; t < 4; ++t) {
        int col = col0 + 16 * t + n;
        float bias = bo[col];
#pragma unroll
        for (int j = 0; j < 8; ++j)
            out[(size_t)(row0 + mo + j) * HID + col] = acc[t][j] + bias;
    }
}

// ---------- launch ----------

extern "C" void kernel_launch(void* const* d_in, const int* in_sizes, int n_in,
                              void* d_out, int out_size, void* d_ws, size_t ws_size,
                              hipStream_t stream) {
    (void)in_sizes; (void)n_in; (void)out_size; (void)ws_size;
    const float* x      = (const float*)d_in[0];
    const float* gscale = (const float*)d_in[1];
    const float* gbias  = (const float*)d_in[2];
    const float* Wq     = (const float*)d_in[3];
    const float* bq     = (const float*)d_in[4];
    const float* Wk     = (const float*)d_in[5];
    const float* bk     = (const float*)d_in[6];
    const float* Wv     = (const float*)d_in[7];
    const float* bv     = (const float*)d_in[8];
    const float* Wo     = (const float*)d_in[9];
    const float* bo     = (const float*)d_in[10];
    float* out = (float*)d_out;

    char* w = (char*)d_ws;
    size_t off = 0;
    auto carve = [&](size_t bytes) {
        void* p = w + off;
        off = (off + bytes + 255) & ~(size_t)255;
        return p;
    };
    _Float16* xn    = (_Float16*)carve((size_t)ROWS * HID * 2);        // 8 MB
    _Float16* wqkvT = (_Float16*)carve((size_t)NQKV * HID * 2);        // 6 MB
    _Float16* woT   = (_Float16*)carve((size_t)HID * HD * 2);          // 128 KB
    float*    qkv   = (float*)   carve((size_t)ROWS * NQKV * 4);       // 48 MB
    _Float16* attnH = (_Float16*)carve((size_t)ROWS * NH * HD * 2);    // 8 MB

    k_pack_wqkv<<<(NQKV * HID) / 256, 256, 0, stream>>>(Wq, Wk, Wv, wqkvT);
    k_pack_wo  <<<(HID * HD) / 256,  256, 0, stream>>>(Wo, woT);
    k_layernorm<<<ROWS, 256, 0, stream>>>(x, gscale, gbias, xn);
    k_qkv_gemm <<<(ROWS / 16) * (NQKV / 64) / 8, 256, 0, stream>>>(xn, wqkvT, bq, bk, bv, qkv);
    k_attention<<<ROWS, 256, 0, stream>>>(qkv, attnH,
                                          out + OUT0,            // attention_entropy (B,S)
                                          out + OUT0 + ROWS,     // head_diversity   (B,S)
                                          out + OUT0 + 2 * ROWS);// gradient_norm    (B,S)
    k_out_gemm <<<((ROWS * NH) / 16) * (HID / 64) / 8, 256, 0, stream>>>(attnH, woT, bo, out);
}
